// ViTAttention_65223373357536
// MI455X (gfx1250) — matrix operand order repaired and checked by reading, compiled, not run on hardware
//
#include <hip/hip_runtime.h>

// ---------------- problem constants ----------------
#define BATCH 4
#define SEQ   2048
#define CDIM  1024
#define HEADS 16
#define HDIM  64
#define BH    (BATCH * HEADS)                 // 64
#define BHSD  ((size_t)BH * SEQ * HDIM)       // 8,388,608 elements per Q/K/V buffer

typedef __bf16 bf16;
typedef __bf16 v8bf  __attribute__((ext_vector_type(8)));
typedef __bf16 v16bf __attribute__((ext_vector_type(16)));
typedef float  v8f   __attribute__((ext_vector_type(8)));
typedef int    v4i   __attribute__((ext_vector_type(4)));

// ---- CDNA5 async global->LDS copy (ASYNCcnt) with portable fallback ----
#if defined(__has_builtin)
#  if __has_builtin(__builtin_amdgcn_global_load_async_to_lds_b128)
#    define USE_ASYNC_LDS 1
#  endif
#endif
#ifndef USE_ASYNC_LDS
#  define USE_ASYNC_LDS 0
#endif

__device__ __forceinline__ void copy16(const bf16* g, bf16* l) {
#if USE_ASYNC_LDS
  __builtin_amdgcn_global_load_async_to_lds_b128(
      (__attribute__((address_space(1))) v4i*)(v4i*)(bf16*)g,
      (__attribute__((address_space(3))) v4i*)(v4i*)l, 0, 0);
#else
  *(uint4*)l = *(const uint4*)g;
#endif
}

__device__ __forceinline__ void wait_async() {
#if USE_ASYNC_LDS
#  if __has_builtin(__builtin_amdgcn_s_wait_asynccnt)
  __builtin_amdgcn_s_wait_asynccnt(0);
#  else
  asm volatile("s_wait_asynccnt 0x0" ::: "memory");
#  endif
#endif
}

__device__ __forceinline__ v8f vzero8() {
  v8f z = {0.f, 0.f, 0.f, 0.f, 0.f, 0.f, 0.f, 0.f};
  return z;
}

// bf16 WMMA: D(16x16,f32) = A(16x32,bf16) * B(32x16,bf16) + C
__device__ __forceinline__ v8f wmma_bf16(v16bf a, v16bf b, v8f c) {
  return __builtin_amdgcn_wmma_f32_16x16x32_bf16(
      false, a, false, b, (short)0, c, false, false);
}

__device__ __forceinline__ v16bf cat16(v8bf l, v8bf h) {
  return __builtin_shufflevector(l, h, 0,1,2,3,4,5,6,7,8,9,10,11,12,13,14,15);
}
// A-fragment (16-bit, 16x32): lane(lo,hi): K chunks at 8*hi and 16+8*hi
__device__ __forceinline__ v16bf afrag(const bf16* p, int hi) {
  v8bf l = *(const v8bf*)(p + 8 * hi);
  v8bf h = *(const v8bf*)(p + 16 + 8 * hi);
  return cat16(l, h);
}
// B-fragment (16-bit, 32x16): lane(lo,hi): K chunks at 8*hi and 16+8*hi
__device__ __forceinline__ v16bf bfrag(const bf16* p, int hi) {
  v8bf l = *(const v8bf*)(p + 8 * hi);
  v8bf h = *(const v8bf*)(p + 16 + 8 * hi);
  return cat16(l, h);
}

// =====================================================================
// elementwise fp32 -> (hi, lo) bf16 split:  x = hi + lo,  lo = bf16(x - hi)
// =====================================================================
__global__ __launch_bounds__(256) void split_kernel(
    const float* __restrict__ x, bf16* __restrict__ hi, bf16* __restrict__ lo)
{
  size_t i = (size_t)blockIdx.x * 256 + threadIdx.x;
  float v = x[i];
  bf16 h = (bf16)v;
  hi[i] = h;
  lo[i] = (bf16)(v - (float)h);
}

// =====================================================================
// bf16x3 GEMM:  out = X(M x K) * W(N x K)^T + bias  with X,W pre-split.
// Block tile 128(M) x 64(N), Kblk 64, double-buffered LDS with async
// global->LDS copies. 8 waves = 4(M) x 2(N); 3-term bf16x3 per 32-K step.
// mode 0: out row-major (M x Ntot);  mode 1: QKV scatter [3][B*H,S,D]
// =====================================================================
#define GST 72   // LDS row stride in bf16 elems: 16B aligned, conflict-free

__global__ __launch_bounds__(256) void gemm_bf16x3_kernel(
    const bf16* __restrict__ Xh, const bf16* __restrict__ Xl,
    const bf16* __restrict__ Wh, const bf16* __restrict__ Wl,
    const float* __restrict__ bias, float* __restrict__ out,
    int K, int Ntot, int mode)
{
  __shared__ bf16 sAh[2][128 * GST], sAl[2][128 * GST];
  __shared__ bf16 sBh[2][64 * GST],  sBl[2][64 * GST];

  const int tid  = threadIdx.x;
  const int lane = tid & 31;
  const int w    = tid >> 5;
  const int lo   = lane & 15;
  const int hi   = lane >> 4;
  const int gm0  = blockIdx.y * 128;
  const int gn0  = blockIdx.x * 64;
  const int mw   = (w & 3) * 32;
  const int nw   = (w >> 2) * 32;

  // stage one K-block (64 cols) of A/B hi+lo into buffer `buf`
  auto stage = [&](int buf, int k0) {
#pragma unroll
    for (int p = 0; p < 4; ++p) {
      int linear = tid + p * 256;          // 1024 16B-chunks per A part
      int row = linear >> 3;
      int c8  = (linear & 7) * 8;
      copy16(Xh + (size_t)(gm0 + row) * K + k0 + c8, &sAh[buf][row * GST + c8]);
      copy16(Xl + (size_t)(gm0 + row) * K + k0 + c8, &sAl[buf][row * GST + c8]);
    }
#pragma unroll
    for (int p = 0; p < 2; ++p) {
      int linear = tid + p * 256;          // 512 16B-chunks per B part
      int row = linear >> 3;
      int c8  = (linear & 7) * 8;
      copy16(Wh + (size_t)(gn0 + row) * K + k0 + c8, &sBh[buf][row * GST + c8]);
      copy16(Wl + (size_t)(gn0 + row) * K + k0 + c8, &sBl[buf][row * GST + c8]);
    }
  };

  v8f acc[2][2];
  acc[0][0] = vzero8(); acc[0][1] = vzero8();
  acc[1][0] = vzero8(); acc[1][1] = vzero8();

  const int NB = K >> 6;
  stage(0, 0);
  wait_async();
  __syncthreads();

  for (int kb = 0; kb < NB; ++kb) {
    const int buf = kb & 1;
    if (kb + 1 < NB) stage(buf ^ 1, (kb + 1) << 6);  // overlap with compute

#pragma unroll
    for (int ks = 0; ks < 2; ++ks) {
      const int kc = ks * 32;
      v16bf ah0 = afrag(&sAh[buf][(mw + lo) * GST + kc], hi);
      v16bf al0 = afrag(&sAl[buf][(mw + lo) * GST + kc], hi);
      v16bf ah1 = afrag(&sAh[buf][(mw + 16 + lo) * GST + kc], hi);
      v16bf al1 = afrag(&sAl[buf][(mw + 16 + lo) * GST + kc], hi);
      v16bf bh0 = bfrag(&sBh[buf][(nw + lo) * GST + kc], hi);
      v16bf bl0 = bfrag(&sBl[buf][(nw + lo) * GST + kc], hi);
      v16bf bh1 = bfrag(&sBh[buf][(nw + 16 + lo) * GST + kc], hi);
      v16bf bl1 = bfrag(&sBl[buf][(nw + 16 + lo) * GST + kc], hi);
      // 3-term bf16x3, interleaved across the 4 accumulators for ILP
      acc[0][0] = wmma_bf16(ah0, bh0, acc[0][0]);
      acc[0][1] = wmma_bf16(ah0, bh1, acc[0][1]);
      acc[1][0] = wmma_bf16(ah1, bh0, acc[1][0]);
      acc[1][1] = wmma_bf16(ah1, bh1, acc[1][1]);
      acc[0][0] = wmma_bf16(ah0, bl0, acc[0][0]);
      acc[0][1] = wmma_bf16(ah0, bl1, acc[0][1]);
      acc[1][0] = wmma_bf16(ah1, bl0, acc[1][0]);
      acc[1][1] = wmma_bf16(ah1, bl1, acc[1][1]);
      acc[0][0] = wmma_bf16(al0, bh0, acc[0][0]);
      acc[0][1] = wmma_bf16(al0, bh1, acc[0][1]);
      acc[1][0] = wmma_bf16(al1, bh0, acc[1][0]);
      acc[1][1] = wmma_bf16(al1, bh1, acc[1][1]);
    }
    wait_async();      // next buffer's copies have landed in LDS
    __syncthreads();
  }

  // ---- epilogue: bias + store (C/D layout: vgpr j -> rows j, j+8) ----
#pragma unroll
  for (int mt = 0; mt < 2; ++mt) {
#pragma unroll
    for (int nt = 0; nt < 2; ++nt) {
      float bb = bias[gn0 + nw + nt * 16 + lo];
#pragma unroll
      for (int j = 0; j < 8; ++j) {
        int r = gm0 + mw + mt * 16 + j + hi * 8;
        int c = gn0 + nw + nt * 16 + lo;
        float val = acc[mt][nt][j] + bb;
        if (mode == 0) {
          out[(size_t)r * Ntot + c] = val;
        } else {
          int t  = c >> 10;
          int hd = c & 1023;
          int b  = r >> 11;
          int s  = r & 2047;
          size_t dst = (size_t)t * BHSD +
                       (((size_t)b * 16 + (hd >> 6)) * 2048 + s) * 64 + (hd & 63);
          out[dst] = val;
        }
      }
    }
  }
}

// =====================================================================
// RoPE + split: reads f32 Q/K ([which][B*H,S,D]), writes rotated values
// as hi/lo bf16. Q additionally pre-scaled by 1/sqrt(D) = 0.125.
// =====================================================================
__global__ __launch_bounds__(256) void rope_split_kernel(
    const float* __restrict__ qkv, const float* __restrict__ cosb,
    const float* __restrict__ sinb,
    bf16* __restrict__ qhi, bf16* __restrict__ qlo,
    bf16* __restrict__ khi, bf16* __restrict__ klo)
{
  size_t idx = (size_t)blockIdx.x * 256 + threadIdx.x;
  int dp = (int)(idx & 31);  size_t rest = idx >> 5;
  int s  = (int)(rest & 2047); rest >>= 11;
  int bh = (int)(rest & 63);
  int which = (int)(rest >> 6);
  int b = bh >> 4;

  const float* p  = qkv + (size_t)which * BHSD + ((size_t)bh * 2048 + s) * 64;
  const float* cp = cosb + ((size_t)b * 2048 + s) * 64;
  const float* sp = sinb + ((size_t)b * 2048 + s) * 64;

  float x0 = p[dp], x1 = p[dp + 32];
  float c0 = cp[dp], c1 = cp[dp + 32];
  float s0 = sp[dp], s1 = sp[dp + 32];
  float scale = which ? 1.0f : 0.125f;
  float y0 = (x0 * c0 - x1 * s0) * scale;   // rotate_half: -x[d+32] for d<32
  float y1 = (x1 * c1 + x0 * s1) * scale;   //  x[d-32]     for d>=32

  bf16* dh = (which ? khi : qhi) + ((size_t)bh * 2048 + s) * 64;
  bf16* dl = (which ? klo : qlo) + ((size_t)bh * 2048 + s) * 64;
  bf16 h0 = (bf16)y0, h1 = (bf16)y1;
  dh[dp]      = h0;  dl[dp]      = (bf16)(y0 - (float)h0);
  dh[dp + 32] = h1;  dl[dp + 32] = (bf16)(y1 - (float)h1);
}

// =====================================================================
// V transpose + split: f32 [B*H,S,D] -> hi/lo bf16 [B*H,D,S]
// =====================================================================
__global__ __launch_bounds__(256) void vsplit_t_kernel(
    const float* __restrict__ v, bf16* __restrict__ vthi, bf16* __restrict__ vtlo)
{
  size_t idx = (size_t)blockIdx.x * 256 + threadIdx.x;
  int s  = (int)(idx & 2047);
  int d  = (int)((idx >> 11) & 63);
  int bh = (int)(idx >> 17);
  float x = v[((size_t)bh * 2048 + s) * 64 + d];
  size_t dst = ((size_t)bh * 64 + d) * 2048 + s;
  bf16 h = (bf16)x;
  vthi[dst] = h;
  vtlo[dst] = (bf16)(x - (float)h);
}

// =====================================================================
// Flash attention (bf16x3). One block per (b,h, 64-query tile), 8 waves:
// wave = (mt 0..3) x (nh 0..1) -> 16 query rows x 32 cols.
// Scores spilled f32 to LDS for online softmax; P split hi/lo bf16 in LDS.
// Output written [B, S, H*D] f32 so the proj GEMM consumes it directly.
// =====================================================================
#define SST 66

__global__ __launch_bounds__(256) void attn_kernel(
    const bf16* __restrict__ Qh, const bf16* __restrict__ Ql,
    const bf16* __restrict__ Kh, const bf16* __restrict__ Kl,
    const bf16* __restrict__ Vth, const bf16* __restrict__ Vtl,
    float* __restrict__ O)
{
  __shared__ float sS[64 * SST];
  __shared__ bf16 sPh[64 * GST], sPl[64 * GST];
  __shared__ float sRed[64 * 4];
  __shared__ float sM[64], sL[64], sAlpha[64];

  const int tid  = threadIdx.x;
  const int lane = tid & 31;
  const int w    = tid >> 5;
  const int lo   = lane & 15;
  const int hi   = lane >> 4;
  const int bh   = blockIdx.x >> 5;
  const int qb   = blockIdx.x & 31;
  const int mt   = w & 3;
  const int nh   = w >> 2;

  // ---- Q fragments (already rotated + scaled), both parts, 2 K-steps ----
  v16bf qfh[2], qfl[2];
  {
    size_t qoff = ((size_t)bh * 2048 + qb * 64 + mt * 16 + lo) * 64;
    const bf16* qrh = Qh + qoff;
    const bf16* qrl = Ql + qoff;
#pragma unroll
    for (int ks = 0; ks < 2; ++ks) {
      qfh[ks] = afrag(qrh + 32 * ks, hi);
      qfl[ks] = afrag(qrl + 32 * ks, hi);
    }
  }
  v8f o0 = vzero8(), o1 = vzero8();

  if (tid < 64) { sM[tid] = -1e30f; sL[tid] = 0.f; }
  __syncthreads();

  for (int kb = 0; kb < 32; ++kb) {
    // ---- scores S = Q K^T (3-term bf16x3) ----
    v8f a0 = vzero8(), a1 = vzero8();
    {
      size_t ko0 = ((size_t)bh * 2048 + kb * 64 + nh * 32 + lo) * 64;
      size_t ko1 = ko0 + 16 * 64;
      const bf16* kr0h = Kh + ko0; const bf16* kr0l = Kl + ko0;
      const bf16* kr1h = Kh + ko1; const bf16* kr1l = Kl + ko1;
#pragma unroll
      for (int ks = 0; ks < 2; ++ks) {
        v16bf b0h = bfrag(kr0h + 32 * ks, hi);
        v16bf b1h = bfrag(kr1h + 32 * ks, hi);
        v16bf b0l = bfrag(kr0l + 32 * ks, hi);
        v16bf b1l = bfrag(kr1l + 32 * ks, hi);
        a0 = wmma_bf16(qfh[ks], b0h, a0);
        a1 = wmma_bf16(qfh[ks], b1h, a1);
        a0 = wmma_bf16(qfh[ks], b0l, a0);
        a1 = wmma_bf16(qfh[ks], b1l, a1);
        a0 = wmma_bf16(qfl[ks], b0h, a0);
        a1 = wmma_bf16(qfl[ks], b1h, a1);
      }
    }
    // ---- spill score tile to LDS (f32) ----
#pragma unroll
    for (int j = 0; j < 8; ++j) {
      int r = mt * 16 + j + hi * 8;
      sS[r * SST + nh * 32 + lo]      = a0[j];
      sS[r * SST + nh * 32 + 16 + lo] = a1[j];
    }
    __syncthreads();

    // ---- online softmax: thread t -> row t/4, 16-col quarter t%4 ----
    {
      const int r = tid >> 2, qq = tid & 3;
      const float* row = &sS[r * SST + qq * 16];
      float mx = -1e30f;
#pragma unroll
      for (int i = 0; i < 16; ++i) mx = fmaxf(mx, row[i]);
      sRed[r * 4 + qq] = mx;
      __syncthreads();
      if (qq == 0) {
        float m_old = sM[r];
        float m4 = fmaxf(fmaxf(sRed[r * 4], sRed[r * 4 + 1]),
                         fmaxf(sRed[r * 4 + 2], sRed[r * 4 + 3]));
        float m_new = fmaxf(m_old, m4);
        sM[r] = m_new;
        sAlpha[r] = __expf(m_old - m_new);
      }
      __syncthreads();
      float m_new = sM[r];
      float sum = 0.f;
      bf16* ph = &sPh[r * GST + qq * 16];
      bf16* pl = &sPl[r * GST + qq * 16];
#pragma unroll
      for (int i = 0; i < 16; ++i) {
        float p = __expf(row[i] - m_new);
        sum += p;
        bf16 hh = (bf16)p;
        ph[i] = hh;
        pl[i] = (bf16)(p - (float)hh);
      }
      sRed[r * 4 + qq] = sum;
      __syncthreads();
      if (qq == 0)
        sL[r] = sL[r] * sAlpha[r] +
                (sRed[r * 4] + sRed[r * 4 + 1] + sRed[r * 4 + 2] + sRed[r * 4 + 3]);
    }

    // ---- rescale running O ----
#pragma unroll
    for (int j = 0; j < 8; ++j) {
      float al = sAlpha[mt * 16 + j + hi * 8];
      o0[j] *= al;
      o1[j] *= al;
    }

    // ---- O += P @ V (P hi/lo from LDS as A-frags, V^T rows as B-frags) ----
    {
      const bf16* prh = &sPh[(mt * 16 + lo) * GST];
      const bf16* prl = &sPl[(mt * 16 + lo) * GST];
      size_t vo0 = ((size_t)bh * 64 + nh * 32 + lo) * 2048 + (size_t)kb * 64;
      size_t vo1 = vo0 + 16 * 2048;
      const bf16* vr0h = Vth + vo0; const bf16* vr0l = Vtl + vo0;
      const bf16* vr1h = Vth + vo1; const bf16* vr1l = Vtl + vo1;
#pragma unroll
      for (int ks = 0; ks < 2; ++ks) {
        v16bf ph = afrag(prh + 32 * ks, hi);
        v16bf pl = afrag(prl + 32 * ks, hi);
        v16bf v0h = bfrag(vr0h + 32 * ks, hi);
        v16bf v1h = bfrag(vr1h + 32 * ks, hi);
        v16bf v0l = bfrag(vr0l + 32 * ks, hi);
        v16bf v1l = bfrag(vr1l + 32 * ks, hi);
        o0 = wmma_bf16(ph, v0h, o0);
        o1 = wmma_bf16(ph, v1h, o1);
        o0 = wmma_bf16(ph, v0l, o0);
        o1 = wmma_bf16(ph, v1l, o1);
        o0 = wmma_bf16(pl, v0h, o0);
        o1 = wmma_bf16(pl, v1h, o1);
      }
    }
    __syncthreads();   // protect sS/sPh/sPl/sRed before next iteration
  }

  // ---- epilogue: divide by row sum, write [B, S, H*D] ----
  const int b = bh >> 4, h = bh & 15;
#pragma unroll
  for (int j = 0; j < 8; ++j) {
    int r = mt * 16 + j + hi * 8;
    float il = 1.0f / sL[r];
    size_t srow = (size_t)b * 2048 + qb * 64 + r;
    float* op = O + srow * 1024 + h * 64 + nh * 32;
    op[lo]      = o0[j] * il;
    op[16 + lo] = o1[j] * il;
  }
}

// =====================================================================
// launcher
// =====================================================================
extern "C" void kernel_launch(void* const* d_in, const int* in_sizes, int n_in,
                              void* d_out, int out_size, void* d_ws, size_t ws_size,
                              hipStream_t stream) {
  (void)in_sizes; (void)n_in; (void)out_size; (void)ws_size;
  const float* hidden = (const float*)d_in[0];
  const float* cosb   = (const float*)d_in[1];
  const float* sinb   = (const float*)d_in[2];
  const float* qkv_w  = (const float*)d_in[3];
  const float* qkv_b  = (const float*)d_in[4];
  const float* proj_w = (const float*)d_in[5];
  const float* proj_b = (const float*)d_in[6];
  float* out = (float*)d_out;

  // ---- workspace layout (with aliasing of the dead qkv-f32 region) ----
  char* base = (char*)d_ws;
  float* qkvf = (float*)base;                          // [3][B*H,S,D] f32
  char* r1 = base + 3 * BHSD * sizeof(float);
  bf16* qhi = (bf16*)r1;
  bf16* qlo = qhi + BHSD;
  bf16* khi = qlo + BHSD;
  bf16* klo = khi + BHSD;
  char* r2 = r1 + 4 * BHSD * sizeof(bf16);
  bf16* vthi = (bf16*)r2;
  bf16* vtlo = vthi + BHSD;
  char* r3 = r2 + 2 * BHSD * sizeof(bf16);
  bf16* xhi = (bf16*)r3;
  bf16* xlo = xhi + BHSD;
  char* r4 = r3 + 2 * BHSD * sizeof(bf16);
  bf16* wqh = (bf16*)r4;
  bf16* wql = wqh + (size_t)3 * CDIM * CDIM;
  char* r5 = r4 + (size_t)2 * 3 * CDIM * CDIM * sizeof(bf16);
  bf16* wph = (bf16*)r5;
  bf16* wpl = wph + (size_t)CDIM * CDIM;
  // after rope/vsplit, qkv-f32 region is dead: reuse it for attn out + split
  float* attnf = qkvf;                                 // [B,S,C] f32
  bf16* ahi = (bf16*)(base + BHSD * sizeof(float));
  bf16* alo = ahi + BHSD;

  dim3 blk(256);
  // 0) precision splits of hidden + weights
  split_kernel<<<BHSD / 256, blk, 0, stream>>>(hidden, xhi, xlo);
  split_kernel<<<(3 * CDIM * CDIM) / 256, blk, 0, stream>>>(qkv_w, wqh, wql);
  split_kernel<<<(CDIM * CDIM) / 256, blk, 0, stream>>>(proj_w, wph, wpl);
  // 1) QKV projection (bf16x3) + scatter to [3][B*H,S,D] f32
  gemm_bf16x3_kernel<<<dim3(3072 / 64, 8192 / 128), blk, 0, stream>>>(
      xhi, xlo, wqh, wql, qkv_b, qkvf, CDIM, 3 * CDIM, /*mode=*/1);
  // 2) RoPE + split Q,K ; transpose + split V
  rope_split_kernel<<<(2u * BH * SEQ * 32) / 256, blk, 0, stream>>>(
      qkvf, cosb, sinb, qhi, qlo, khi, klo);
  vsplit_t_kernel<<<BHSD / 256, blk, 0, stream>>>(qkvf + 2 * BHSD, vthi, vtlo);
  // 3) flash attention (bf16x3) -> [B,S,C] f32
  attn_kernel<<<BH * (SEQ / 64), blk, 0, stream>>>(
      qhi, qlo, khi, klo, vthi, vtlo, attnf);
  // 4) split attention output, then output projection (bf16x3) -> d_out
  split_kernel<<<BHSD / 256, blk, 0, stream>>>(attnf, ahi, alo);
  gemm_bf16x3_kernel<<<dim3(1024 / 64, 8192 / 128), blk, 0, stream>>>(
      ahi, alo, wph, wpl, proj_b, out, CDIM, CDIM, /*mode=*/0);
}
